// InstanceHead_20272245637463
// MI455X (gfx1250) — matrix operand order repaired and checked by reading, not compiled, not run on hardware
//
#include <hip/hip_runtime.h>

typedef _Float16 half_t;
typedef __attribute__((ext_vector_type(16))) _Float16 v16h;
typedef __attribute__((ext_vector_type(8)))  _Float16 v8h;
typedef __attribute__((ext_vector_type(8)))  float    v8f;
typedef __attribute__((ext_vector_type(4)))  float    v4f;

#define HASH_BITS 19
#define HASH_SIZE (1u << HASH_BITS)
#define HASH_MASK (HASH_SIZE - 1u)
#define TAU_F 0.1f
#define TOPK 128
#define L_DIM 96
#define D_DIM 32

// ---------- CDNA5 async copy to LDS (ASYNCcnt-tracked, ISA 15.18.3 op 98) ----------
__device__ __forceinline__ void async_b128_to_lds(void* lds_dst, const void* gsrc) {
  unsigned lds_addr = (unsigned)(unsigned long long)lds_dst;  // generic LDS ptr: addr[31:0] = LDS byte address
  asm volatile("global_load_async_to_lds_b128 %0, %1, off"
               :: "v"(lds_addr), "v"(gsrc) : "memory");
}
__device__ __forceinline__ void wait_asynccnt0() {
  asm volatile("s_wait_asynccnt 0x0" ::: "memory");
}

// ---------- hashing (jax sphash analogue: injective coord key + open addressing) ----------
__device__ __forceinline__ long long encode4(int b, int x, int y, int z) {
  long long c = (long long)b;
  c = c * 2048 + (x + 512);
  c = c * 2048 + (y + 512);
  c = c * 2048 + (z + 512);
  return c;
}
__device__ __forceinline__ unsigned hash64(long long kk) {
  unsigned long long k = (unsigned long long)kk;
  k ^= k >> 33; k *= 0xff51afd7ed558ccdULL;
  k ^= k >> 33; k *= 0xc4ceb9fe1a85ec53ULL;
  k ^= k >> 33;
  return (unsigned)k & HASH_MASK;
}
__device__ __forceinline__ int ht_insert(long long* keys, long long key) {
  unsigned s = hash64(key);
  for (;;) {
    unsigned long long prev = atomicCAS((unsigned long long*)(keys + s),
                                        0xFFFFFFFFFFFFFFFFull, (unsigned long long)key);
    if (prev == 0xFFFFFFFFFFFFFFFFull || prev == (unsigned long long)key) return (int)s;
    s = (s + 1) & HASH_MASK;
  }
}
__device__ __forceinline__ int ht_lookup(const long long* keys, long long key) {
  unsigned s = hash64(key);
  for (;;) {
    long long k = keys[s];
    if (k == key) return (int)s;
    if (k == -1ll) return -1;
    s = (s + 1) & HASH_MASK;
  }
}

// ---------- prep: Wt[n][k] = (f16) W[k][n]  (B fragments become contiguous 32B loads) ----------
__global__ void wt_kernel(const float* __restrict__ W, half_t* __restrict__ Wt) {
  int i = blockIdx.x * blockDim.x + threadIdx.x;
  if (i >= L_DIM * D_DIM) return;
  int k = i / D_DIM, n = i % D_DIM;
  Wt[(size_t)n * L_DIM + k] = (half_t)W[i];
}

// ---------- Stage A: vdesc = relu(feats @ W), f16 WMMA, K=96 in 3 steps ----------
__global__ __launch_bounds__(256) void vdesc_kernel(const float* __restrict__ feats,
                                                    const half_t* __restrict__ Wt,
                                                    half_t* __restrict__ vdesc, int N) {
  __shared__ __align__(32) half_t sW[D_DIM * L_DIM];  // 6 KB, async-staged once per block
  const int tid = threadIdx.x;
  for (int o = tid * 16; o < D_DIM * L_DIM * (int)sizeof(half_t); o += 256 * 16)
    async_b128_to_lds((char*)sW + o, (const char*)Wt + o);
  wait_asynccnt0();
  __syncthreads();

  const int wave = tid >> 5;
  const int lane = tid & 31;
  const int rowBase = (blockIdx.x * 4 + (wave >> 1)) * 16;
  const int colTile = wave & 1;
  if (rowBase >= N) return;
  const int mrow  = lane & 15;
  const int khalf = lane >> 4;

  const float* arow0 = feats + (size_t)(rowBase + mrow) * L_DIM;
  __builtin_prefetch(arow0 + (size_t)64 * L_DIM, 0, 1);  // next block's row stream

  v8f acc = {};
  for (int ks = 0; ks < 3; ++ks) {
    const float* arow = arow0 + ks * 32;
    v4f p0 = *(const v4f*)(arow + khalf * 16);
    v4f p1 = *(const v4f*)(arow + khalf * 16 + 4);
    v4f p2 = *(const v4f*)(arow + khalf * 16 + 8);
    v4f p3 = *(const v4f*)(arow + khalf * 16 + 12);
    v16h a;
#pragma unroll
    for (int i = 0; i < 4; ++i) {
      a[i]      = (half_t)p0[i];
      a[4 + i]  = (half_t)p1[i];
      a[8 + i]  = (half_t)p2[i];
      a[12 + i] = (half_t)p3[i];
    }
    v16h b = *(const v16h*)(sW + (size_t)(colTile * 16 + mrow) * L_DIM + ks * 32 + khalf * 16);
    acc = __builtin_amdgcn_wmma_f32_16x16x32_f16(false, a, false, b, (short)0, acc, false, false);
  }
#pragma unroll
  for (int r = 0; r < 8; ++r) {
    float v = acc[r] > 0.f ? acc[r] : 0.f;
    vdesc[(size_t)(rowBase + r + 8 * khalf) * D_DIM + colTile * 16 + mrow] = (half_t)v;
  }
}

// ---------- Stage B: revoxelize (unique clusters via hash) ----------
__global__ void insert_keys_kernel(const int* __restrict__ coords, const float* __restrict__ offsets,
                                   long long* __restrict__ h1k, int N) {
  int i = blockIdx.x * blockDim.x + threadIdx.x;
  if (i >= N) return;
  int b = coords[i * 4 + 0];
  int x = coords[i * 4 + 1] + (int)offsets[i * 3 + 0];
  int y = coords[i * 4 + 2] + (int)offsets[i * 3 + 1];
  int z = coords[i * 4 + 3] + (int)offsets[i * 3 + 2];
  ht_insert(h1k, encode4(b, x, y, z));
}

__global__ void enumerate_kernel(const long long* __restrict__ h1k, unsigned* __restrict__ h1v,
                                 unsigned* __restrict__ counter) {
  int s = blockIdx.x * blockDim.x + threadIdx.x;
  if (s >= (int)HASH_SIZE) return;
  if (h1k[s] != -1ll) h1v[s] = atomicAdd(counter, 1u);
}

__global__ void assign_kernel(const int* __restrict__ coords, const float* __restrict__ offsets,
                              const long long* __restrict__ h1k, const unsigned* __restrict__ h1v,
                              int* __restrict__ inv, int* __restrict__ counts,
                              float* __restrict__ csum, int N) {
  int i = blockIdx.x * blockDim.x + threadIdx.x;
  if (i >= N) return;
  int b = coords[i * 4 + 0];
  int x = coords[i * 4 + 1] + (int)offsets[i * 3 + 0];
  int y = coords[i * 4 + 2] + (int)offsets[i * 3 + 1];
  int z = coords[i * 4 + 3] + (int)offsets[i * 3 + 2];
  int slot = ht_lookup(h1k, encode4(b, x, y, z));
  int c = (int)h1v[slot];
  inv[i] = c;
  atomicAdd(&counts[c], 1);
  atomicAdd(&csum[c * 4 + 0], (float)b);
  atomicAdd(&csum[c * 4 + 1], (float)x);
  atomicAdd(&csum[c * 4 + 2], (float)y);
  atomicAdd(&csum[c * 4 + 3], (float)z);
}

__global__ void cluster_coords_kernel(const long long* __restrict__ h1k, const unsigned* __restrict__ h1v,
                                      const int* __restrict__ counts, const float* __restrict__ csum,
                                      int* __restrict__ occ, long long* __restrict__ h2k,
                                      unsigned* __restrict__ h2v) {
  int s = blockIdx.x * blockDim.x + threadIdx.x;
  if (s >= (int)HASH_SIZE) return;
  if (h1k[s] == -1ll) return;
  int c = (int)h1v[s];
  float cf = (float)counts[c];
  int b = (int)(csum[c * 4 + 0] / cf);   // trunc-toward-zero == astype(int32)
  int x = (int)(csum[c * 4 + 1] / cf);
  int y = (int)(csum[c * 4 + 2] / cf);
  int z = (int)(csum[c * 4 + 3] / cf);
  occ[c * 4 + 0] = b; occ[c * 4 + 1] = x; occ[c * 4 + 2] = y; occ[c * 4 + 3] = z;
  int slot = ht_insert(h2k, encode4(b, x, y, z));
  atomicMin(&h2v[slot], (unsigned)c);
}

__global__ void scatter_kernel(const float* __restrict__ feats, const int* __restrict__ inv,
                               const int* __restrict__ counts, float* __restrict__ csums96, int N) {
  long long idx = (long long)blockIdx.x * blockDim.x + threadIdx.x;
  if (idx >= (long long)N * L_DIM) return;
  int i = (int)(idx / L_DIM);
  int t = (int)(idx % L_DIM);
  int c = inv[i];
  float cnt = (float)counts[c];
  atomicAdd(&csums96[(size_t)c * L_DIM + t], feats[(size_t)i * L_DIM + t] / cnt);
}

// ---------- Stage C: peak finding ----------
__global__ void peak_hash_kernel(const int* __restrict__ coords, const float* __restrict__ scores,
                                 long long* __restrict__ h3k, unsigned* __restrict__ h3v, int N) {
  int i = blockIdx.x * blockDim.x + threadIdx.x;
  if (i >= N) return;
  if (scores[i] <= TAU_F) return;
  int slot = ht_insert(h3k, encode4(coords[i * 4 + 0], coords[i * 4 + 1],
                                    coords[i * 4 + 2], coords[i * 4 + 3]));
  atomicMin(&h3v[slot], (unsigned)i);   // representative = lowest original index (stable-sort tie)
}

__global__ void cand_kernel(const int* __restrict__ coords, const float* __restrict__ scores,
                            const long long* __restrict__ h3k, const unsigned* __restrict__ h3v,
                            unsigned long long* __restrict__ cand, int N) {
  int i = blockIdx.x * blockDim.x + threadIdx.x;
  if (i >= N) return;
  float sc = scores[i];
  unsigned long long packed = 0;
  if (sc > TAU_F) {
    int b = coords[i * 4 + 0], x = coords[i * 4 + 1], y = coords[i * 4 + 2], z = coords[i * 4 + 3];
    float hm = -__builtin_inff();
    for (int dx = -1; dx <= 1; ++dx)
      for (int dy = -1; dy <= 1; ++dy)
        for (int dz = -1; dz <= 1; ++dz) {
          int slot = ht_lookup(h3k, encode4(b, x + dx, y + dy, z + dz));
          if (slot >= 0) {
            unsigned rep = h3v[slot];
            float rs = scores[rep];
            hm = rs > hm ? rs : hm;
          }
        }
    if (hm == sc) {  // local maximum -> candidate; larger score wins, tie -> smaller index
      packed = ((unsigned long long)__float_as_uint(sc) << 32) | (unsigned)(~(unsigned)i);
    }
  }
  cand[i] = packed;
}

__global__ __launch_bounds__(256) void topk_kernel(unsigned long long* __restrict__ cand, int N,
                                                   unsigned long long* __restrict__ top,
                                                   float* __restrict__ conf) {
  __shared__ unsigned long long sv[256];
  for (int round = 0; round < TOPK; ++round) {
    unsigned long long best = 0;
    for (int i = threadIdx.x; i < N; i += blockDim.x) {
      unsigned long long v = cand[i];
      best = v > best ? v : best;
    }
    sv[threadIdx.x] = best;
    __syncthreads();
    for (int off = 128; off > 0; off >>= 1) {
      if ((int)threadIdx.x < off) {
        unsigned long long o = sv[threadIdx.x + off];
        if (o > sv[threadIdx.x]) sv[threadIdx.x] = o;
      }
      __syncthreads();
    }
    unsigned long long w = sv[0];
    if (threadIdx.x == 0) {
      top[round]  = w;
      conf[round] = (w != 0) ? __uint_as_float((unsigned)(w >> 32)) : 0.f;
      if (w != 0) cand[~(unsigned)(w & 0xFFFFFFFFull)] = 0;  // remove winner
    }
    __syncthreads();
  }
}

// ---------- 27-tap avg pool over cluster voxels, only for the 128 peaks ----------
__global__ void peakfeat_kernel(const unsigned long long* __restrict__ top, const int* __restrict__ inv,
                                const int* __restrict__ occ, const long long* __restrict__ h2k,
                                const unsigned* __restrict__ h2v, const float* __restrict__ csums96,
                                float* __restrict__ pfeats) {
  int k = blockIdx.x;
  int t = threadIdx.x;  // 0..95
  unsigned long long w = top[k];
  float val = 0.f;
  if (w != 0) {
    int i = (int)(~(unsigned)(w & 0xFFFFFFFFull));
    int c = inv[i];
    int b = occ[c * 4 + 0], x = occ[c * 4 + 1], y = occ[c * 4 + 2], z = occ[c * 4 + 3];
    float acc = 0.f; int cnt = 0;
    for (int dx = -1; dx <= 1; ++dx)
      for (int dy = -1; dy <= 1; ++dy)
        for (int dz = -1; dz <= 1; ++dz) {
          int slot = ht_lookup(h2k, encode4(b, x + dx, y + dy, z + dz));
          if (slot >= 0) {
            unsigned nc = h2v[slot];
            acc += csums96[(size_t)nc * L_DIM + t];
            cnt++;
          }
        }
    val = acc / (float)(cnt > 0 ? cnt : 1);
  }
  pfeats[(size_t)k * L_DIM + t] = val;
}

// ---------- Stage D: cdesc = [bg ; conf * relu(pfeats @ W)] as f16 (144x32, zero-padded) ----------
__global__ __launch_bounds__(256) void cdesc_kernel(const float* __restrict__ pfeats,
                                                    const half_t* __restrict__ Wt,
                                                    const float* __restrict__ conf,
                                                    const float* __restrict__ bg,
                                                    half_t* __restrict__ cdesc) {
  int tid = threadIdx.x;
  for (int i = tid; i < 15 * D_DIM; i += 256) cdesc[129 * D_DIM + i] = (half_t)0.f;  // pad rows
  if (tid < D_DIM) cdesc[tid] = (half_t)bg[tid];                                     // row 0 = bg
  const int wave = tid >> 5;
  const int lane = tid & 31;
  const int mrow = lane & 15;
  const int khalf = lane >> 4;
  const int rowBase = wave * 16;   // 8 waves x 16 rows = 128 peak rows
  for (int ct = 0; ct < 2; ++ct) {
    v8f acc = {};
    for (int ks = 0; ks < 3; ++ks) {
      const float* arow = pfeats + (size_t)(rowBase + mrow) * L_DIM + ks * 32;
      v4f p0 = *(const v4f*)(arow + khalf * 16);
      v4f p1 = *(const v4f*)(arow + khalf * 16 + 4);
      v4f p2 = *(const v4f*)(arow + khalf * 16 + 8);
      v4f p3 = *(const v4f*)(arow + khalf * 16 + 12);
      v16h a;
#pragma unroll
      for (int i = 0; i < 4; ++i) {
        a[i]      = (half_t)p0[i];
        a[4 + i]  = (half_t)p1[i];
        a[8 + i]  = (half_t)p2[i];
        a[12 + i] = (half_t)p3[i];
      }
      v16h b = *(const v16h*)(Wt + (size_t)(ct * 16 + mrow) * L_DIM + ks * 32 + khalf * 16);
      acc = __builtin_amdgcn_wmma_f32_16x16x32_f16(false, a, false, b, (short)0, acc, false, false);
    }
#pragma unroll
    for (int r = 0; r < 8; ++r) {
      int row = rowBase + r + 8 * khalf;
      float v = acc[r] > 0.f ? acc[r] : 0.f;
      cdesc[(size_t)(1 + row) * D_DIM + ct * 16 + mrow] = (half_t)(v * conf[row]);
    }
  }
}

// ---------- Stage E: inst = vdesc @ cdesc.T (N x 129), single-K-step WMMA, B staged in LDS ----------
__global__ __launch_bounds__(256) void inst_kernel(const half_t* __restrict__ vdesc,
                                                   const half_t* __restrict__ cdesc,
                                                   float* __restrict__ inst, int N) {
  __shared__ __align__(32) half_t sB[144 * D_DIM];  // 9216 B, async-staged once per block
  const int tid = threadIdx.x;
  for (int o = tid * 16; o < 144 * D_DIM * (int)sizeof(half_t); o += 256 * 16)
    async_b128_to_lds((char*)sB + o, (const char*)cdesc + o);
  wait_asynccnt0();
  __syncthreads();

  const int wave = tid >> 5;
  const int lane = tid & 31;
  const int rowBase = (blockIdx.x * 8 + wave) * 16;
  if (rowBase >= N) return;
  const int mrow = lane & 15;
  const int khalf = lane >> 4;

  const half_t* arow = vdesc + (size_t)(rowBase + mrow) * D_DIM;
  v8h lo = *(const v8h*)(arow + khalf * 16);
  v8h hi = *(const v8h*)(arow + khalf * 16 + 8);
  v16h a = __builtin_shufflevector(lo, hi, 0, 1, 2, 3, 4, 5, 6, 7, 8, 9, 10, 11, 12, 13, 14, 15);

  for (int ct = 0; ct < 9; ++ct) {
    v16h b = *(const v16h*)(sB + (size_t)(ct * 16 + mrow) * D_DIM + khalf * 16);
    v8f acc = {};
    acc = __builtin_amdgcn_wmma_f32_16x16x32_f16(false, a, false, b, (short)0, acc, false, false);
    int col = ct * 16 + mrow;
    if (col < 129) {
#pragma unroll
      for (int r = 0; r < 8; ++r)
        inst[(size_t)(rowBase + r + 8 * khalf) * 129 + col] = acc[r];
    }
  }
}

extern "C" void kernel_launch(void* const* d_in, const int* in_sizes, int n_in,
                              void* d_out, int out_size, void* d_ws, size_t ws_size,
                              hipStream_t stream) {
  const int*   coords  = (const int*)  d_in[0];
  const float* feats   = (const float*)d_in[1];
  const float* scores  = (const float*)d_in[2];
  const float* offsets = (const float*)d_in[3];
  const float* W       = (const float*)d_in[4];
  const float* bg      = (const float*)d_in[5];
  const int N = in_sizes[0] / 4;

  char* p = (char*)d_ws;
  auto take = [&](size_t bytes) -> char* {
    char* r = p; p += (bytes + 255) & ~(size_t)255; return r;
  };
  half_t*             vdesc   = (half_t*)   take((size_t)N * D_DIM * sizeof(half_t));
  long long*          h1k     = (long long*)take((size_t)HASH_SIZE * 8);
  unsigned*           h1v     = (unsigned*) take((size_t)HASH_SIZE * 4);
  long long*          h2k     = (long long*)take((size_t)HASH_SIZE * 8);
  unsigned*           h2v     = (unsigned*) take((size_t)HASH_SIZE * 4);
  long long*          h3k     = (long long*)take((size_t)HASH_SIZE * 8);
  unsigned*           h3v     = (unsigned*) take((size_t)HASH_SIZE * 4);
  int*                inv     = (int*)      take((size_t)N * 4);
  int*                counts  = (int*)      take((size_t)N * 4);
  float*              csum    = (float*)    take((size_t)N * 16);
  int*                occ     = (int*)      take((size_t)N * 16);
  float*              csums96 = (float*)    take((size_t)N * L_DIM * 4);
  unsigned long long* cand    = (unsigned long long*)take((size_t)N * 8);
  unsigned long long* top     = (unsigned long long*)take(TOPK * 8);
  float*              pfeats  = (float*)    take((size_t)TOPK * L_DIM * 4);
  half_t*             cdesc   = (half_t*)   take((size_t)144 * D_DIM * sizeof(half_t));
  half_t*             Wt      = (half_t*)   take((size_t)D_DIM * L_DIM * sizeof(half_t));
  unsigned*           counter = (unsigned*) take(256);

  float* conf = (float*)d_out;          // (128,)
  float* inst = (float*)d_out + TOPK;   // (N,129)

  hipMemsetAsync(h1k, 0xFF, (size_t)HASH_SIZE * 8, stream);
  hipMemsetAsync(h2k, 0xFF, (size_t)HASH_SIZE * 8, stream);
  hipMemsetAsync(h3k, 0xFF, (size_t)HASH_SIZE * 8, stream);
  hipMemsetAsync(h2v, 0xFF, (size_t)HASH_SIZE * 4, stream);
  hipMemsetAsync(h3v, 0xFF, (size_t)HASH_SIZE * 4, stream);
  hipMemsetAsync(counts, 0, (size_t)N * 4, stream);
  hipMemsetAsync(csum, 0, (size_t)N * 16, stream);
  hipMemsetAsync(csums96, 0, (size_t)N * L_DIM * 4, stream);
  hipMemsetAsync(counter, 0, 4, stream);

  const int gN = (N + 255) / 256;
  const int gH = (int)((HASH_SIZE + 255) / 256);

  wt_kernel<<<(L_DIM * D_DIM + 255) / 256, 256, 0, stream>>>(W, Wt);
  vdesc_kernel<<<(N + 63) / 64, 256, 0, stream>>>(feats, Wt, vdesc, N);
  insert_keys_kernel<<<gN, 256, 0, stream>>>(coords, offsets, h1k, N);
  enumerate_kernel<<<gH, 256, 0, stream>>>(h1k, h1v, counter);
  assign_kernel<<<gN, 256, 0, stream>>>(coords, offsets, h1k, h1v, inv, counts, csum, N);
  cluster_coords_kernel<<<gH, 256, 0, stream>>>(h1k, h1v, counts, csum, occ, h2k, h2v);
  scatter_kernel<<<(int)(((long long)N * L_DIM + 255) / 256), 256, 0, stream>>>(feats, inv, counts, csums96, N);
  peak_hash_kernel<<<gN, 256, 0, stream>>>(coords, scores, h3k, h3v, N);
  cand_kernel<<<gN, 256, 0, stream>>>(coords, scores, h3k, h3v, cand, N);
  topk_kernel<<<1, 256, 0, stream>>>(cand, N, top, conf);
  peakfeat_kernel<<<TOPK, L_DIM, 0, stream>>>(top, inv, occ, h2k, h2v, csums96, pfeats);
  cdesc_kernel<<<1, 256, 0, stream>>>(pfeats, Wt, conf, bg, cdesc);
  inst_kernel<<<(N + 127) / 128, 256, 0, stream>>>(vdesc, cdesc, inst, N);
}
